// _GraphMessageLayer_61203283968403
// MI455X (gfx1250) — matrix operand order repaired and checked by reading, not compiled, not run on hardware
//
#include <hip/hip_runtime.h>
#include <math.h>

// ---------------------------------------------------------------------------
// GraphMessageLayer for MI455X (gfx1250), bf16 WMMA (f32 accumulate)
//   N=20000 nodes, E=320000 edges, D=256
// Pipeline:
//   prep:  weights f32 -> bf16 transposed [N][K]; degree histogram
//   edges: X = gather(node[src]) + edge  -> GEMM(W1m)+GELU -> GEMM(W2m)
//          -> atomicAdd scatter into agg (= d_out, zeroed per call)
//   nodes: A = agg/deg -> GEMM(W1u)+GELU -> GEMM(W2u) + b + residual
//          -> LayerNorm -> d_out (in place over agg rows owned by the block)
// ---------------------------------------------------------------------------

#define DD      256
#define DD2     512
#define NNODES  20000
#define NEDGES  320000

typedef __attribute__((ext_vector_type(16))) __bf16 v16bf;
typedef __attribute__((ext_vector_type(8)))  float  v8f;

union FragU { uint4 u[2]; v16bf v; };

__device__ __forceinline__ unsigned short f2bf(float f) {
  unsigned int u = __float_as_uint(f);
  u += 0x7FFFu + ((u >> 16) & 1u);          // round-to-nearest-even
  return (unsigned short)(u >> 16);
}

__device__ __forceinline__ float gelu_exact(float x) {
  return 0.5f * x * (1.0f + erff(x * 0.70710678118654752440f));
}

// One 16x16 f32 output tile:  A (LDS, row-major [m][k], stride srA bf16 elems)
//                           x B (global bf16, transposed [n][KDIM])
// A layout (ISA 7.12.2, 16-bit A 16x32): lane<16 -> K chunks {0..7},{16..23};
// lane>=16 -> {8..15},{24..31}.  B layout: same K chunks as A.
template <int KDIM>
__device__ __forceinline__ v8f gemm_tile(const unsigned short* __restrict__ Abase,
                                         int srA,
                                         const unsigned short* __restrict__ BT,
                                         int m0, int n0, int lane) {
  v8f c = {};
  const int l15  = lane & 15;
  const int aoff = (lane < 16) ? 0 : 8;
  const int boff = (lane < 16) ? 0 : 8;
  const unsigned short* arow = Abase + (m0 + l15) * srA;
  const unsigned short* brow = BT + (long long)(n0 + l15) * KDIM;
#pragma unroll
  for (int k0 = 0; k0 < KDIM; k0 += 32) {
    FragU a, b;
    a.u[0] = *(const uint4*)(arow + k0 + aoff);
    a.u[1] = *(const uint4*)(arow + k0 + 16 + aoff);
    b.u[0] = *(const uint4*)(brow + k0 + boff);
    b.u[1] = *(const uint4*)(brow + k0 + 16 + boff);
    c = __builtin_amdgcn_wmma_f32_16x16x32_bf16(false, a.v, false, b.v,
                                                (short)0, c, false, false);
  }
  return c;
}

// --------------------------- prep kernels ----------------------------------

// f32 W[K][Nn]  ->  bf16 WT[Nn][K]
__global__ void k_cvt_t(const float* __restrict__ W, unsigned short* __restrict__ WT,
                        int K, int Nn) {
  int i = blockIdx.x * blockDim.x + threadIdx.x;
  if (i < K * Nn) {
    int k = i / Nn, n = i % Nn;
    WT[(long long)n * K + k] = f2bf(W[i]);
  }
}

__global__ void k_deg(const long long* __restrict__ dstIdx, float* __restrict__ deg) {
  int e = blockIdx.x * blockDim.x + threadIdx.x;
  if (e < NEDGES) atomicAdd(&deg[(int)dstIdx[e]], 1.0f);
}

// --------------------------- edge kernel -----------------------------------

#define SRX 264   // 256 + 8 bf16 pad: per-lane b128 row reads spread across banks

__global__ __launch_bounds__(256)
void k_edges(const float* __restrict__ nodeEmb,
             const long long* __restrict__ eidx,      // [2][E]
             const float* __restrict__ edgeEmb,
             const unsigned short* __restrict__ W1T, const float* __restrict__ b1,
             const unsigned short* __restrict__ W2T, const float* __restrict__ b2,
             float* __restrict__ agg) {
  __shared__ __align__(16) unsigned short Xs[128 * SRX];
  __shared__ __align__(16) unsigned short Hs[128 * SRX];
  const int tid   = threadIdx.x;
  const int eBase = blockIdx.x * 128;
  const long long* srcIdx = eidx;
  const long long* dstIdx = eidx + NEDGES;

  // Stage A: gather node[src] + edge, convert to bf16 (col = tid, one row/iter)
  for (int r = 0; r < 128; ++r) {
    long long s = srcIdx[eBase + r];
    float v = nodeEmb[s * DD + tid] + edgeEmb[(long long)(eBase + r) * DD + tid];
    Xs[r * SRX + tid] = f2bf(v);
  }
  __syncthreads();

  const int w = tid >> 5, lane = tid & 31;
  const int l15  = lane & 15;
  const int mrow = 16 * w + ((lane < 16) ? 0 : 8);

  // GEMM1 + GELU -> Hs   (wave w owns M-tile w; loops all 16 N-tiles)
  for (int nt = 0; nt < 16; ++nt) {
    v8f c = gemm_tile<256>(Xs, SRX, W1T, 16 * w, nt * 16, lane);
    int n = nt * 16 + l15;
    float bias = b1[n];
#pragma unroll
    for (int r = 0; r < 8; ++r)
      Hs[(mrow + r) * SRX + n] = f2bf(gelu_exact(c[r] + bias));
  }
  __syncthreads();

  // GEMM2 + bias -> atomic scatter-add into agg[dst]
  for (int nt = 0; nt < 16; ++nt) {
    v8f c = gemm_tile<256>(Hs, SRX, W2T, 16 * w, nt * 16, lane);
    int n = nt * 16 + l15;
    float bias = b2[n];
#pragma unroll
    for (int r = 0; r < 8; ++r) {
      long long dn = dstIdx[eBase + mrow + r];
      atomicAdd(&agg[dn * DD + n], c[r] + bias);
    }
  }
}

// --------------------------- node kernel -----------------------------------

#define SRH 520   // 512 + 8 pad

__global__ __launch_bounds__(256)
void k_nodes(const float* __restrict__ nodeEmb,
             const float* __restrict__ deg,
             const unsigned short* __restrict__ W1T, const float* __restrict__ b1, // [512][256]
             const unsigned short* __restrict__ W2T, const float* __restrict__ b2, // [256][512]
             const float* __restrict__ gamma, const float* __restrict__ beta,
             float* __restrict__ out /* holds agg on entry; final result on exit */) {
  __shared__ __align__(16) unsigned short Xs[64 * SRX];
  __shared__ __align__(16) unsigned short Hs[64 * SRH];
  __shared__ float Rs[64][260];
  const int tid   = threadIdx.x;
  const int nBase = blockIdx.x * 64;

  // Stage A: agg / max(deg,1) -> bf16 LDS
  for (int r = 0; r < 64; ++r) {
    int node = nBase + r;
    float v = 0.f;
    if (node < NNODES) {
      float dg = deg[node];
      v = out[(long long)node * DD + tid] / fmaxf(dg, 1.0f);
    }
    Xs[r * SRX + tid] = f2bf(v);
  }
  __syncthreads();

  const int w = tid >> 5, lane = tid & 31;
  const int l15 = lane & 15;
  const int mt  = w & 3;                     // M-tile 0..3 (64 rows)
  const int mrow = 16 * mt + ((lane < 16) ? 0 : 8);

  // GEMM1: [64,256] x [256,512] + GELU -> Hs
  const int ntBase1 = (w >> 2) * 16;         // waves 0-3: N-tiles 0..15, 4-7: 16..31
  for (int i = 0; i < 16; ++i) {
    int n0 = (ntBase1 + i) * 16;
    v8f c = gemm_tile<256>(Xs, SRX, W1T, 16 * mt, n0, lane);
    int n = n0 + l15;
    float bias = b1[n];
#pragma unroll
    for (int r = 0; r < 8; ++r)
      Hs[(mrow + r) * SRH + n] = f2bf(gelu_exact(c[r] + bias));
  }
  __syncthreads();

  // GEMM2: [64,512] x [512,256] + bias + residual -> Rs (f32)
  const int ntBase2 = (w >> 2) * 8;
  for (int i = 0; i < 8; ++i) {
    int n0 = (ntBase2 + i) * 16;
    v8f c = gemm_tile<512>(Hs, SRH, W2T, 16 * mt, n0, lane);
    int n = n0 + l15;
    float bias = b2[n];
#pragma unroll
    for (int r = 0; r < 8; ++r) {
      int node = nBase + mrow + r;
      float res = (node < NNODES) ? nodeEmb[(long long)node * DD + n] : 0.f;
      Rs[mrow + r][n] = c[r] + bias + res;
    }
  }
  __syncthreads();

  // Stage D: LayerNorm, one thread per row (LN cost is negligible)
  if (tid < 64) {
    int node = nBase + tid;
    if (node < NNODES) {
      float mu = 0.f;
      for (int c = 0; c < DD; ++c) mu += Rs[tid][c];
      mu *= (1.0f / DD);
      float var = 0.f;
      for (int c = 0; c < DD; ++c) { float d = Rs[tid][c] - mu; var += d * d; }
      var *= (1.0f / DD);
      float rstd = rsqrtf(var + 1e-5f);
      for (int c = 0; c < DD; ++c)
        out[(long long)node * DD + c] = (Rs[tid][c] - mu) * rstd * gamma[c] + beta[c];
    }
  }
}

// --------------------------- launch ----------------------------------------

extern "C" void kernel_launch(void* const* d_in, const int* in_sizes, int n_in,
                              void* d_out, int out_size, void* d_ws, size_t ws_size,
                              hipStream_t stream) {
  (void)in_sizes; (void)n_in; (void)out_size; (void)ws_size;

  const float*     nodeEmb = (const float*)d_in[0];
  const long long* eidx    = (const long long*)d_in[1];   // int64 [2,E]
  const float*     edgeEmb = (const float*)d_in[2];
  const float*     W1m = (const float*)d_in[3];
  const float*     b1m = (const float*)d_in[4];
  const float*     W2m = (const float*)d_in[5];
  const float*     b2m = (const float*)d_in[6];
  const float*     W1u = (const float*)d_in[7];
  const float*     b1u = (const float*)d_in[8];
  const float*     W2u = (const float*)d_in[9];
  const float*     b2u = (const float*)d_in[10];
  const float*     gamma = (const float*)d_in[11];
  const float*     beta  = (const float*)d_in[12];
  float* out = (float*)d_out;

  // workspace layout (~0.85 MB): bf16 transposed weights + degree
  char* ws = (char*)d_ws;
  unsigned short* W1mT = (unsigned short*)(ws);                       // 256*256
  unsigned short* W2mT = (unsigned short*)(ws + 131072);              // 256*256
  unsigned short* W1uT = (unsigned short*)(ws + 262144);              // [512][256]
  unsigned short* W2uT = (unsigned short*)(ws + 524288);              // [256][512]
  float*          degp = (float*)(ws + 786432);                       // 20000 f32

  hipMemsetAsync(d_out, 0, (size_t)NNODES * DD * sizeof(float), stream);
  hipMemsetAsync(degp, 0, (size_t)NNODES * sizeof(float), stream);

  k_cvt_t<<<(DD * DD  + 255) / 256, 256, 0, stream>>>(W1m, W1mT, DD,  DD);
  k_cvt_t<<<(DD * DD  + 255) / 256, 256, 0, stream>>>(W2m, W2mT, DD,  DD);
  k_cvt_t<<<(DD * DD2 + 255) / 256, 256, 0, stream>>>(W1u, W1uT, DD,  DD2);
  k_cvt_t<<<(DD2 * DD + 255) / 256, 256, 0, stream>>>(W2u, W2uT, DD2, DD);

  k_deg<<<(NEDGES + 255) / 256, 256, 0, stream>>>(eidx + NEDGES, degp);

  k_edges<<<NEDGES / 128, 256, 0, stream>>>(nodeEmb, eidx, edgeEmb,
                                            W1mT, b1m, W2mT, b2m, out);

  k_nodes<<<(NNODES + 63) / 64, 256, 0, stream>>>(nodeEmb, degp,
                                                  W1uT, b1u, W2uT, b2u,
                                                  gamma, beta, out);
}
